// LocalAggregator_28561532518857
// MI455X (gfx1250) — matrix operand order repaired and checked by reading, not compiled, not run on hardware
//
#include <hip/hip_runtime.h>
#include <math.h>

typedef _Float16 v16h __attribute__((ext_vector_type(16)));
typedef _Float16 v8h  __attribute__((ext_vector_type(8)));
typedef _Float16 v2h  __attribute__((ext_vector_type(2)));
typedef float    v8f  __attribute__((ext_vector_type(8)));
typedef float    v4f  __attribute__((ext_vector_type(4)));
typedef float    v2f  __attribute__((ext_vector_type(2)));

#define NN 256
#define DD 256
#define NEG_MASK (-9.0e15f)
#define SLOPE 0.2f

// ---- dynamic LDS layout (bytes) ----
#define HS_OFF  0         // [256][256] f16 row-major (XOR-swizzled within row)
#define HST_OFF 131072    // [256][256] f16 transposed (XOR-swizzled within row)
#define AS_OFF  262144    // [4][256]   f16 attention vectors (linear; broadcast reads)
#define LG_OFF  264192    // [16][256]  f32 logits -> alpha (XOR-swizzled within row)
#define OT_OFF  280576    // [16][256]  f32 raw aggregated rows (linear)
#define SMEM_BYTES 296960

// ---- XOR bank swizzles (16B granularity; pure permutation within a row) ----
// f16 rows of 256: 32 groups of 8 halves
__device__ __forceinline__ int swH(int row, int d) {
  return ((((d >> 3) ^ row) & 31) << 3) | (d & 7);
}
__device__ __forceinline__ int swHg(int row, int grp) {  // group-granular
  return ((grp ^ row) & 31) << 3;
}
// f32 rows of 256: 64 groups of 4 floats (XOR low 4 bits with row)
__device__ __forceinline__ int swL(int r, int c) {
  return ((((c >> 2) ^ r) & 15) << 2) | ((c >> 2) & ~15) * 4 + (c & 3);
}
__device__ __forceinline__ int swLg(int r, int grp) {
  return (((grp ^ r) & 15) | (grp & ~15)) << 2;
}

__device__ __forceinline__ v16h joinh(v8h lo, v8h hi) {
  v16h r;
#pragma unroll
  for (int i = 0; i < 8; ++i) { r[i] = lo[i]; r[i + 8] = hi[i]; }
  return r;
}

__device__ __forceinline__ float wave_max32(float v) {
#pragma unroll
  for (int o = 16; o > 0; o >>= 1) v = fmaxf(v, __shfl_xor(v, o, 32));
  return v;
}
__device__ __forceinline__ float wave_sum32(float v) {
#pragma unroll
  for (int o = 16; o > 0; o >>= 1) v += __shfl_xor(v, o, 32);
  return v;
}

__global__ __launch_bounds__(256, 1)
void gat_fused_kernel(const float* __restrict__ hidden,
                      const int*   __restrict__ adj,
                      const float* __restrict__ avec,
                      float* __restrict__ out) {
  extern __shared__ char smem[];
  _Float16* Hs  = (_Float16*)(smem + HS_OFF);
  _Float16* HsT = (_Float16*)(smem + HST_OFF);
  _Float16* aS  = (_Float16*)(smem + AS_OFF);
  float*    Lg  = (float*)(smem + LG_OFF);
  float*    Ot  = (float*)(smem + OT_OFF);

  const int tid  = threadIdx.x;
  const int lane = tid & 31;
  const int wid  = tid >> 5;
  const int half = lane >> 4;     // 16-lane group selector (WMMA layouts)
  const int l16  = lane & 15;
  const int i0   = blockIdx.x * 16;   // row tile
  const int b    = blockIdx.y;        // batch

  const float* Hb = hidden + (size_t)b * NN * DD;

  // ---------- Phase 0: stage H_b as f16 into LDS (row-major + transposed) ----------
  // 2 consecutive elements of one row per iteration: coalesced 8B global loads,
  // swizzled b32 store to Hs (conflict-free), 2 swizzled b16 stores to HsT (~2-way).
#pragma unroll 1
  for (int q = tid; q < (NN * DD) / 2; q += 256) {
    const int row = q >> 7;
    const int col = (q & 127) * 2;
    v2f x = *(const v2f*)(Hb + row * DD + col);
    v2h hx; hx[0] = (_Float16)x[0]; hx[1] = (_Float16)x[1];
    *(v2h*)&Hs[row * DD + swH(row, col)] = hx;      // col,col+1 share a 16B group
    HsT[col * NN + swH(col, row)]           = hx[0];
    HsT[(col + 1) * NN + swH(col + 1, row)] = hx[1];
  }
  for (int q = tid; q < 4 * DD; q += 256) aS[q] = (_Float16)avec[q];
  __syncthreads();

  // ---------- Phase 1: logits[16 x 256] = select_k(leaky(H diag(a_k) H^T)) ----------
  {
    const int arow = i0 + l16;
    v8f acc[2][4];
#pragma unroll
    for (int t = 0; t < 2; ++t)
#pragma unroll
      for (int k = 0; k < 4; ++k) acc[t][k] = v8f{};

#pragma unroll 1
    for (int kk = 0; kk < 8; ++kk) {
      const int kb = kk * 32;
      const int g0 = kk * 4;
      // A fragment (lane = row): swizzled groups
      v8h a_lo = *(const v8h*)&Hs[arow * DD + swHg(arow, g0 + half)];
      v8h a_hi = *(const v8h*)&Hs[arow * DD + swHg(arow, g0 + 2 + half)];
      v16h A = joinh(a_lo, a_hi);
      v16h As[4];
#pragma unroll
      for (int k = 0; k < 4; ++k) {
        // a_k is uniform across the 16-lane group -> broadcast reads, keep linear
        v8h s_lo = *(const v8h*)&aS[k * DD + kb + half * 8];
        v8h s_hi = *(const v8h*)&aS[k * DD + kb + 16 + half * 8];
        As[k] = A * joinh(s_lo, s_hi);
      }
#pragma unroll
      for (int t = 0; t < 2; ++t) {
        const int bcol = (wid + 8 * t) * 16 + l16;
        v8h b_lo = *(const v8h*)&Hs[bcol * DD + swHg(bcol, g0 + half)];
        v8h b_hi = *(const v8h*)&Hs[bcol * DD + swHg(bcol, g0 + 2 + half)];
        v16h Bf = joinh(b_lo, b_hi);
#pragma unroll
        for (int k = 0; k < 4; ++k)
          acc[t][k] = __builtin_amdgcn_wmma_f32_16x16x32_f16(
              false, As[k], false, Bf, (short)0, acc[t][k], false, false);
      }
    }

    // leaky relu + branchless adjacency select, write logits (swizzled) to LDS
#pragma unroll
    for (int t = 0; t < 2; ++t) {
      const int bcol = (wid + 8 * t) * 16 + l16;
      int av[8];
#pragma unroll
      for (int g = 0; g < 8; ++g)
        av[g] = adj[((size_t)b * NN + i0 + g + 8 * half) * NN + bcol];
#pragma unroll
      for (int g = 0; g < 8; ++g) {
        const int rr = g + 8 * half;
        const float e0 = fmaxf(acc[t][0][g], SLOPE * acc[t][0][g]);
        const float e1 = fmaxf(acc[t][1][g], SLOPE * acc[t][1][g]);
        const float e2 = fmaxf(acc[t][2][g], SLOPE * acc[t][2][g]);
        const float e3 = fmaxf(acc[t][3][g], SLOPE * acc[t][3][g]);
        float lv = NEG_MASK;
        lv = (av[g] == 1) ? e0 : lv;
        lv = (av[g] == 2) ? e1 : lv;
        lv = (av[g] == 3) ? e2 : lv;
        lv = (av[g] == 4) ? e3 : lv;
        Lg[rr * NN + swL(rr, bcol)] = lv;
      }
    }
  }
  __syncthreads();

  // ---------- Phase 2: softmax over each of the 16 rows (2 rows per wave) ----------
  // Reads the row as a group-permutation (max/sum are order independent).
#pragma unroll 1
  for (int rr = 0; rr < 2; ++rr) {
    const int r = wid * 2 + rr;
    float* Lrow = Lg + r * NN;
    const int o0 = swLg(r, lane);        // groups 0..31
    const int o1 = swLg(r, 32 + lane);   // groups 32..63
    v4f x0 = *(const v4f*)(Lrow + o0);
    v4f x1 = *(const v4f*)(Lrow + o1);
    float m = x0[0];
#pragma unroll
    for (int u = 1; u < 4; ++u) m = fmaxf(m, x0[u]);
#pragma unroll
    for (int u = 0; u < 4; ++u) m = fmaxf(m, x1[u]);
    m = wave_max32(m);
    v4f e0, e1; float s = 0.f;
#pragma unroll
    for (int u = 0; u < 4; ++u) { e0[u] = __expf(x0[u] - m); s += e0[u]; }
#pragma unroll
    for (int u = 0; u < 4; ++u) { e1[u] = __expf(x1[u] - m); s += e1[u]; }
    s = wave_sum32(s);
    const float inv = 1.f / s;
#pragma unroll
    for (int u = 0; u < 4; ++u) { e0[u] *= inv; e1[u] *= inv; }
    *(v4f*)(Lrow + o0) = e0;
    *(v4f*)(Lrow + o1) = e1;
  }
  __syncthreads();

  // ---------- Phase 3: Ot[16 x 256] = alpha x H_b via WMMA (transposed H) ----------
#pragma unroll 1
  for (int ct = wid; ct < 16; ct += 8) {
    const int ccol = ct * 16 + l16;
    v8f acc = {};
#pragma unroll 1
    for (int jj = 0; jj < 8; ++jj) {
      const int gl = jj * 8;   // f32 group base within an Lg row
      // A = alpha (f32 LDS -> f16 fragment); lane = row -> swizzled groups
      v4f f0 = *(const v4f*)&Lg[l16 * NN + swLg(l16, gl + 2 * half)];
      v4f f1 = *(const v4f*)&Lg[l16 * NN + swLg(l16, gl + 2 * half + 1)];
      v4f f2 = *(const v4f*)&Lg[l16 * NN + swLg(l16, gl + 4 + 2 * half)];
      v4f f3 = *(const v4f*)&Lg[l16 * NN + swLg(l16, gl + 4 + 2 * half + 1)];
      v16h Af;
#pragma unroll
      for (int u = 0; u < 4; ++u) {
        Af[u]      = (_Float16)f0[u];
        Af[u + 4]  = (_Float16)f1[u];
        Af[u + 8]  = (_Float16)f2[u];
        Af[u + 12] = (_Float16)f3[u];
      }
      // B[j, c] = h[j][c] from transposed copy; lane = row -> swizzled groups
      v8h b_lo = *(const v8h*)&HsT[ccol * NN + swHg(ccol, jj * 4 + half)];
      v8h b_hi = *(const v8h*)&HsT[ccol * NN + swHg(ccol, jj * 4 + 2 + half)];
      v16h Bf = joinh(b_lo, b_hi);
      acc = __builtin_amdgcn_wmma_f32_16x16x32_f16(
          false, Af, false, Bf, (short)0, acc, false, false);
    }
#pragma unroll
    for (int g = 0; g < 8; ++g)
      Ot[(g + 8 * half) * NN + ccol] = acc[g];   // consecutive lanes -> no conflict
  }
  __syncthreads();

  // ---------- Phase 4: L2 normalize rows + residual, store ----------
#pragma unroll 1
  for (int rr = 0; rr < 2; ++rr) {
    const int r = wid * 2 + rr;
    const float* Orow = Ot + r * NN;
    v4f x0 = *(const v4f*)(Orow + lane * 4);          // bank-perfect sweeps
    v4f x1 = *(const v4f*)(Orow + 128 + lane * 4);
    float ss = 0.f;
#pragma unroll
    for (int u = 0; u < 4; ++u) ss += x0[u] * x0[u] + x1[u] * x1[u];
    ss = wave_sum32(ss);
    const float scale = 1.f / fmaxf(sqrtf(ss), 1e-12f);
    const size_t gb = ((size_t)b * NN + (i0 + r)) * DD + lane * 4;
    v4f h0 = *(const v4f*)(hidden + gb);
    v4f h1 = *(const v4f*)(hidden + gb + 128);
    v4f o0, o1;
#pragma unroll
    for (int u = 0; u < 4; ++u) {
      o0[u] = x0[u] * scale + h0[u];
      o1[u] = x1[u] * scale + h1[u];
    }
    *(v4f*)(out + gb)       = o0;
    *(v4f*)(out + gb + 128) = o1;
  }
}

extern "C" void kernel_launch(void* const* d_in, const int* in_sizes, int n_in,
                              void* d_out, int out_size, void* d_ws, size_t ws_size,
                              hipStream_t stream) {
  (void)in_sizes; (void)n_in; (void)d_ws; (void)ws_size; (void)out_size;
  const float* hidden = (const float*)d_in[0];
  const int*   adjm   = (const int*)d_in[1];
  const float* avec   = (const float*)d_in[2];
  float* out = (float*)d_out;

  // allow the large dynamic-LDS request (CDNA5: up to 320 KB per workgroup)
  (void)hipFuncSetAttribute((const void*)gat_fused_kernel,
                            hipFuncAttributeMaxDynamicSharedMemorySize,
                            SMEM_BYTES);

  dim3 grid(16, 64);   // 16 row-tiles x B=64 batches = 1024 workgroups
  gat_fused_kernel<<<grid, 256, SMEM_BYTES, stream>>>(hidden, adjm, avec, out);
}
